// VGAE_4483945857666
// MI455X (gfx1250) — compile-verified, matrix operand order checked by reading, not run on hardware
//
#include <hip/hip_runtime.h>

// ---------------------------------------------------------------------------
// VGAE forward for MI455X (gfx1250, wave32, WMMA).
// All GEMMs use v_wmma_f32_16x16x32_bf16 (f32 accumulate).
// Dominant cost is the 604 MB ZZt store (~26 us at 23.3 TB/s); Z/X/W stay
// L2-resident (192 MB L2), so GEMMs are pure WMMA + streaming stores.
// ---------------------------------------------------------------------------

typedef __bf16 bf16;
typedef __attribute__((ext_vector_type(8)))  __bf16 v8bf;
typedef __attribute__((ext_vector_type(16))) __bf16 v16bf;
typedef __attribute__((ext_vector_type(8)))  float  v8f;

#define NODES 12288
#define EDGES 393216
#define D_IN  512
#define D_H   256
#define D_Z   64

// ---------------- helpers --------------------------------------------------

__global__ void zero_f32(float* __restrict__ p, long long n) {
    long long i = (long long)blockIdx.x * blockDim.x + threadIdx.x;
    if (i < n) p[i] = 0.0f;
}

__global__ void cast_bf16(const float* __restrict__ s, bf16* __restrict__ d, long long n) {
    long long i = (long long)blockIdx.x * blockDim.x + threadIdx.x;
    if (i < n) d[i] = (bf16)s[i];
}

// s is [R,C] row-major f32 -> d is [C,R] row-major bf16 (i.e. transposed)
__global__ void cast_transpose_bf16(const float* __restrict__ s, bf16* __restrict__ d,
                                    int R, int C) {
    long long i = (long long)blockIdx.x * blockDim.x + threadIdx.x;
    if (i >= (long long)R * C) return;
    int r = (int)(i / C);
    int c = (int)(i - (long long)r * C);
    d[(size_t)c * R + r] = (bf16)s[i];
}

__global__ void count_deg(const long long* __restrict__ dst, float* __restrict__ deg, int e) {
    int i = blockIdx.x * blockDim.x + threadIdx.x;
    if (i < e) atomicAdd(deg + (int)dst[i], 1.0f);
}

__global__ void calc_norm(const float* __restrict__ deg, float* __restrict__ nrm, int n) {
    int i = blockIdx.x * blockDim.x + threadIdx.x;
    if (i < n) nrm[i] = rsqrtf(deg[i] + 1.0f);
}

// ---------------- WMMA GEMM: C[M,Nc] = A[M,K](bf16) @ BT[Nc,K](bf16)^T + bias
// One wave per 16x16 output tile; fragments loaded as two b128s per lane.

__device__ __forceinline__ v16bf load_frag(const bf16* __restrict__ p, int ld,
                                           int row0, int k0) {
    int lane = threadIdx.x & 31;
    int h    = lane >> 4;           // half-wave selects K sub-block
    int r    = row0 + (lane & 15);
    const bf16* base = p + (size_t)r * ld + k0 + 8 * h;
    v8bf lo = *reinterpret_cast<const v8bf*>(base);        // K = 8h .. 8h+7
    v8bf hi = *reinterpret_cast<const v8bf*>(base + 16);   // K = 16+8h .. 16+8h+7
    return __builtin_shufflevector(lo, hi, 0, 1, 2, 3, 4, 5, 6, 7,
                                            8, 9, 10, 11, 12, 13, 14, 15);
}

__global__ void __launch_bounds__(256)
gemm_bf16_wmma(const bf16* __restrict__ A, const bf16* __restrict__ BT,
               const float* __restrict__ bias, float* __restrict__ C,
               int K, int Nc, int tiles_x) {
    int wave = blockIdx.x * (blockDim.x >> 5) + (threadIdx.x >> 5);
    int ty = wave / tiles_x;
    int tx = wave - ty * tiles_x;
    int row0 = ty << 4;
    int col0 = tx << 4;

    v8f acc = {};
    for (int k0 = 0; k0 < K; k0 += 32) {
        v16bf a = load_frag(A,  K, row0, k0);
        v16bf b = load_frag(BT, K, col0, k0);
        acc = __builtin_amdgcn_wmma_f32_16x16x32_bf16(
                  false, a, false, b, (short)0, acc, false, false);
    }

    int lane = threadIdx.x & 31;
    int h    = lane >> 4;
    int col  = col0 + (lane & 15);
    float bv = bias ? bias[col] : 0.0f;
#pragma unroll
    for (int v = 0; v < 8; ++v) {
        int row = row0 + v + 8 * h;
        C[(size_t)row * Nc + col] = acc[v] + bv;
    }
}

// ---------------- edge-parallel normalized scatter-add ----------------------
// agg[dst, f] += h[src, f] * norm[src]; F is a power of two -> shifts only.

template <int F>
__global__ void agg_scatter(const long long* __restrict__ src,
                            const long long* __restrict__ dst,
                            const float* __restrict__ h,
                            const float* __restrict__ nrm,
                            float* __restrict__ agg, int e) {
    long long total  = (long long)e * F;
    long long stride = (long long)gridDim.x * blockDim.x;
    for (long long i = (long long)blockIdx.x * blockDim.x + threadIdx.x;
         i < total; i += stride) {
        int ed = (int)(i / F);
        int f  = (int)(i - (long long)ed * F);
        int s  = (int)src[ed];
        int d  = (int)dst[ed];
        atomicAdd(agg + (size_t)d * F + f, h[(size_t)s * F + f] * nrm[s]);
    }
}

// H = relu(norm*(agg + norm*H0)); in-place into H0, plus bf16 copy
__global__ void post_conv1(float* __restrict__ H0, const float* __restrict__ agg,
                           const float* __restrict__ nrm, bf16* __restrict__ Hb,
                           long long total) {
    long long i = (long long)blockIdx.x * blockDim.x + threadIdx.x;
    if (i >= total) return;
    int node = (int)(i >> 8);                       // F = 256
    float nv = nrm[node];
    float v  = nv * (agg[i] + nv * H0[i]);
    v = v > 0.0f ? v : 0.0f;
    H0[i] = v;
    Hb[i] = (bf16)v;
}

// Hsig[i] = H[i,:] . Wsig + bsig   (K = 256, one thread per row)
__global__ void gemv_sig(const float* __restrict__ H, const float* __restrict__ Wsig,
                         const float* __restrict__ bsig, float* __restrict__ Hsig, int n) {
    int i = blockIdx.x * blockDim.x + threadIdx.x;
    if (i >= n) return;
    const float* row = H + (size_t)i * D_H;
    float acc = 0.0f;
#pragma unroll 8
    for (int k = 0; k < D_H; ++k) acc += row[k] * Wsig[k];
    Hsig[i] = acc + bsig[0];
}

// mus/logsigma2s outputs + reparameterized Z (bf16) for the decoder
__global__ void post_conv2(const float* __restrict__ Hmu, const float* __restrict__ aggmu,
                           const float* __restrict__ Hsig, const float* __restrict__ aggsig,
                           const float* __restrict__ nrm, const float* __restrict__ eps,
                           float* __restrict__ mus_out, float* __restrict__ lsig_out,
                           bf16* __restrict__ Zb, int n) {
    long long i = (long long)blockIdx.x * blockDim.x + threadIdx.x;
    if (i >= (long long)n * D_Z) return;
    int node = (int)(i >> 6);                       // F = 64
    int z    = (int)(i & 63);
    float nv = nrm[node];
    float ls = nv * (aggsig[node] + nv * Hsig[node]);
    if (z == 0) lsig_out[node] = ls;
    float sg = __expf(0.5f * ls);
    float mu = nv * (aggmu[i] + nv * Hmu[i]);
    mus_out[i] = mu;
    Zb[i] = (bf16)(mu + sg * eps[i]);
}

// ---------------------------------------------------------------------------

extern "C" void kernel_launch(void* const* d_in, const int* in_sizes, int n_in,
                              void* d_out, int out_size, void* d_ws, size_t ws_size,
                              hipStream_t stream) {
    const float*     X    = (const float*)d_in[0];
    const long long* ei   = (const long long*)d_in[1];   // int64 per reference
    const float*     W1   = (const float*)d_in[2];
    const float*     b1   = (const float*)d_in[3];
    const float*     Wmu  = (const float*)d_in[4];
    const float*     bmu  = (const float*)d_in[5];
    const float*     Wsig = (const float*)d_in[6];
    const float*     bsig = (const float*)d_in[7];
    const float*     eps  = (const float*)d_in[8];

    const long long* e_src = ei;
    const long long* e_dst = ei + EDGES;

    float* zzt      = (float*)d_out;                                   // [N,N]
    float* mus_out  = zzt + (size_t)NODES * NODES;                     // [N,64]
    float* lsig_out = mus_out + (size_t)NODES * D_Z;                   // [N]

    // ---- carve workspace ----
    size_t off = 0;
    char*  base = (char*)d_ws;
    auto carve = [&](size_t bytes) -> void* {
        void* p = base + off;
        off = (off + bytes + 255) & ~(size_t)255;
        return p;
    };
    float* deg    = (float*)carve((size_t)NODES * 4);
    float* nrm    = (float*)carve((size_t)NODES * 4);
    bf16*  Xb     = (bf16*) carve((size_t)NODES * D_IN * 2);
    bf16*  W1Tb   = (bf16*) carve((size_t)D_H * D_IN * 2);       // W1^T
    bf16*  WmuTb  = (bf16*) carve((size_t)D_Z * D_H * 2);        // Wmu^T
    float* H0     = (float*)carve((size_t)NODES * D_H * 4);      // then relu'd H
    float* aggH   = (float*)carve((size_t)NODES * D_H * 4);
    bf16*  Hb     = (bf16*) carve((size_t)NODES * D_H * 2);
    float* Hmu    = (float*)carve((size_t)NODES * D_Z * 4);
    float* aggmu  = (float*)carve((size_t)NODES * D_Z * 4);
    float* Hsig   = (float*)carve((size_t)NODES * 4);
    float* aggsig = (float*)carve((size_t)NODES * 4);
    bf16*  Zb     = (bf16*) carve((size_t)NODES * D_Z * 2);
    (void)ws_size; (void)in_sizes; (void)n_in; (void)out_size;

    const int T = 256;
    auto blks = [](long long n) { return (unsigned)((n + 255) / 256); };

    // ---- zero accumulators ----
    zero_f32<<<blks(NODES), T, 0, stream>>>(deg, NODES);
    zero_f32<<<blks((long long)NODES * D_H), T, 0, stream>>>(aggH, (long long)NODES * D_H);
    zero_f32<<<blks((long long)NODES * D_Z), T, 0, stream>>>(aggmu, (long long)NODES * D_Z);
    zero_f32<<<blks(NODES), T, 0, stream>>>(aggsig, NODES);

    // ---- bf16 casts (weights pre-transposed for contiguous B-fragments) ----
    cast_bf16<<<blks((long long)NODES * D_IN), T, 0, stream>>>(X, Xb, (long long)NODES * D_IN);
    cast_transpose_bf16<<<blks((long long)D_IN * D_H), T, 0, stream>>>(W1, W1Tb, D_IN, D_H);
    cast_transpose_bf16<<<blks((long long)D_H * D_Z), T, 0, stream>>>(Wmu, WmuTb, D_H, D_Z);

    // ---- degree + symmetric norm ----
    count_deg<<<blks(EDGES), T, 0, stream>>>(e_dst, deg, EDGES);
    calc_norm<<<blks(NODES), T, 0, stream>>>(deg, nrm, NODES);

    // ---- GCN layer 1: H0 = X @ W1 + b1  (WMMA, M=12288 K=512 Nc=256) ----
    {
        int tiles_x = D_H / 16, tiles_y = NODES / 16;
        gemm_bf16_wmma<<<(unsigned)(tiles_x * tiles_y / 8), T, 0, stream>>>(
            Xb, W1Tb, b1, H0, D_IN, D_H, tiles_x);
    }
    agg_scatter<D_H><<<8192, T, 0, stream>>>(e_src, e_dst, H0, nrm, aggH, EDGES);
    post_conv1<<<blks((long long)NODES * D_H), T, 0, stream>>>(
        H0, aggH, nrm, Hb, (long long)NODES * D_H);

    // ---- GCN layer 2 (mu head): Hmu = H @ Wmu + bmu  (M=12288 K=256 Nc=64) ----
    {
        int tiles_x = D_Z / 16, tiles_y = NODES / 16;
        gemm_bf16_wmma<<<(unsigned)(tiles_x * tiles_y / 8), T, 0, stream>>>(
            Hb, WmuTb, bmu, Hmu, D_H, D_Z, tiles_x);
    }
    // ---- sigma head: Hsig = H @ Wsig + bsig  (Nc=1, plain GEMV) ----
    gemv_sig<<<blks(NODES), T, 0, stream>>>(H0, Wsig, bsig, Hsig, NODES);

    agg_scatter<D_Z><<<4096, T, 0, stream>>>(e_src, e_dst, Hmu, nrm, aggmu, EDGES);
    agg_scatter<1>  <<<512,  T, 0, stream>>>(e_src, e_dst, Hsig, nrm, aggsig, EDGES);

    // ---- reparameterize: writes mus / logsigma2s outputs + Z(bf16) ----
    post_conv2<<<blks((long long)NODES * D_Z), T, 0, stream>>>(
        Hmu, aggmu, Hsig, aggsig, nrm, eps, mus_out, lsig_out, Zb, NODES);

    // ---- decoder: ZZt = Z @ Z^T  (M=Nc=12288, K=64; BT == Z itself) ----
    {
        int tiles_x = NODES / 16, tiles_y = NODES / 16;
        gemm_bf16_wmma<<<(unsigned)((long long)tiles_x * tiles_y / 8), T, 0, stream>>>(
            Zb, Zb, nullptr, zzt, D_Z, NODES, tiles_x);
    }
}
